// Moondream3Attention_8804682956999
// MI455X (gfx1250) — compile-verified, matrix operand order checked by reading, not run on hardware
//
#include <hip/hip_runtime.h>
#include <hip/hip_bf16.h>
#include <cstdint>

#define S_LEN 2048
#define DMODEL 2048
#define NH 32
#define NKV 8
#define DHEAD 64

typedef __attribute__((ext_vector_type(16))) __bf16 v16bf;
typedef __attribute__((ext_vector_type(8)))  float  v8f;

union ABf {
  v16bf v;
  unsigned short s[16];
  unsigned int   u[8];
};

__device__ __forceinline__ unsigned short f32_to_bf16(float f) {
  unsigned int x = __float_as_uint(f);
  unsigned int r = x + 0x7FFFu + ((x >> 16) & 1u);
  return (unsigned short)(r >> 16);
}

// A/B bf16 fragment index helpers (ISA 16-bit A-matrix 16x32 layout):
// lane<16 -> kbase 0, lane>=16 -> kbase 8; element j -> K = j + kbase + (j<8?0:8)
__device__ __forceinline__ int pair_k(int r, int kb) {
  return ((r < 4) ? (2 * r) : (2 * r + 8)) + kb;
}
__device__ __forceinline__ int elem_k(int j, int kb) {
  return j + kb + ((j < 8) ? 0 : 8);
}

__device__ __forceinline__ v8f wmma_bf16(const ABf& a, const ABf& b, v8f c) {
  return __builtin_amdgcn_wmma_f32_16x16x32_bf16(false, a.v, false, b.v,
                                                 (short)0, c, false, false);
}

// ds_swizzle xor-exchange (group-of-32 mode: and=31, or=0, xor=IMM>>10).
// Stays within 16-lane halves for masks 1/2/4/8.
template <int IMM>
__device__ __forceinline__ float swz(float x) {
  return __int_as_float(__builtin_amdgcn_ds_swizzle(__float_as_int(x), IMM));
}
__device__ __forceinline__ float half_max16(float v) {
  v = fmaxf(v, swz<0x041F>(v));   // xor 1
  v = fmaxf(v, swz<0x081F>(v));   // xor 2
  v = fmaxf(v, swz<0x101F>(v));   // xor 4
  v = fmaxf(v, swz<0x201F>(v));   // xor 8
  return v;
}
__device__ __forceinline__ float half_sum16(float v) {
  v += swz<0x041F>(v);
  v += swz<0x081F>(v);
  v += swz<0x101F>(v);
  v += swz<0x201F>(v);
  return v;
}

// ---------------------------------------------------------------------------
// Tiled GEMM: C[M,N] f32 = A[M,K] (f32 or bf16) * B[K,N] (f32, up to 3 column
// -partitioned weight pointers). Block tile 128x128x32, 256 threads (8 waves),
// each wave computes 64x32 via 4x2 grid of 16x16 WMMA tiles.
// ---------------------------------------------------------------------------
template <bool A_IS_BF16>
__global__ __launch_bounds__(256) void gemm_wmma(
    const void* __restrict__ Aptr,
    const float* __restrict__ B0, const float* __restrict__ B1,
    const float* __restrict__ B2,
    float* __restrict__ C,
    int M, int N, int K, int n0, int n1, int ldb0, int ldb1, int ldb2)
{
  __shared__ alignas(16) unsigned short sA[128 * 32];   // [m][k] bf16
  __shared__ alignas(16) unsigned short sB[128 * 32];   // [n][k] bf16 (transposed)

  const int tid  = threadIdx.x;
  const int lane = tid & 31;
  const int wid  = tid >> 5;
  const int wm   = wid >> 2;        // 0..1
  const int wn   = wid & 3;         // 0..3
  const int half = lane >> 4;
  const int ml   = lane & 15;
  const int kb   = half * 8;

  const int gm0 = blockIdx.x * 128;
  const int gn0 = blockIdx.y * 128;

  const float*          Af = (const float*)Aptr;
  const unsigned short* Ab = (const unsigned short*)Aptr;

  v8f acc[4][2];
#pragma unroll
  for (int i = 0; i < 4; ++i)
#pragma unroll
    for (int j = 0; j < 2; ++j)
#pragma unroll
      for (int r = 0; r < 8; ++r) acc[i][j][r] = 0.f;

  const int nk = K >> 5;
  for (int kt = 0; kt < nk; ++kt) {
    const int k0 = kt << 5;
    // stage A tile 128x32 (convert f32->bf16 on the fly)
#pragma unroll
    for (int i = 0; i < 16; ++i) {
      int idx = i * 256 + tid;
      int row = idx >> 5, col = idx & 31;
      if (A_IS_BF16) sA[idx] = Ab[(size_t)(gm0 + row) * K + k0 + col];
      else           sA[idx] = f32_to_bf16(Af[(size_t)(gm0 + row) * K + k0 + col]);
    }
    // stage B tile 32x128 transposed into [n][k]
#pragma unroll
    for (int i = 0; i < 16; ++i) {
      int idx = i * 256 + tid;
      int r = idx >> 7, c = idx & 127;
      int gc = gn0 + c;
      const float* bp; int ld, cl;
      if (gc < n0)      { bp = B0; ld = ldb0; cl = gc; }
      else if (gc < n1) { bp = B1; ld = ldb1; cl = gc - n0; }
      else              { bp = B2; ld = ldb2; cl = gc - n1; }
      sB[c * 32 + r] = f32_to_bf16(bp[(size_t)(k0 + r) * ld + cl]);
    }
    __syncthreads();

    if (kt + 1 < nk) {
      size_t eb = A_IS_BF16 ? 2 : 4;
      __builtin_prefetch((const char*)Aptr +
                         ((size_t)(gm0 + (tid & 127)) * K + k0 + 32) * eb, 0, 1);
    }

    ABf afr[4], bfr[2];
#pragma unroll
    for (int mt = 0; mt < 4; ++mt) {
      int row = wm * 64 + mt * 16 + ml;
#pragma unroll
      for (int r = 0; r < 8; ++r)
        afr[mt].u[r] = *(const unsigned int*)&sA[row * 32 + pair_k(r, kb)];
    }
#pragma unroll
    for (int nt = 0; nt < 2; ++nt) {
      int coln = wn * 32 + nt * 16 + ml;
#pragma unroll
      for (int r = 0; r < 8; ++r)
        bfr[nt].u[r] = *(const unsigned int*)&sB[coln * 32 + pair_k(r, kb)];
    }
#pragma unroll
    for (int mt = 0; mt < 4; ++mt)
#pragma unroll
      for (int nt = 0; nt < 2; ++nt)
        acc[mt][nt] = wmma_bf16(afr[mt], bfr[nt], acc[mt][nt]);
    __syncthreads();
  }

  // write C (C/D layout: VGPR r -> M = r + 8*half, N = lane&15)
#pragma unroll
  for (int mt = 0; mt < 4; ++mt)
#pragma unroll
    for (int nt = 0; nt < 2; ++nt)
#pragma unroll
      for (int r = 0; r < 8; ++r) {
        int row = gm0 + wm * 64 + mt * 16 + r + 8 * half;
        int col = gn0 + wn * 32 + nt * 16 + ml;
        C[(size_t)row * N + col] = acc[mt][nt][r];
      }
}

// ---------------------------------------------------------------------------
// tau projections: feat = gelu(qkv row), tok = tanh(feat @ tau_W), plus
// positional tau. One block per sequence position.
// ---------------------------------------------------------------------------
__global__ __launch_bounds__(256) void tau_kernel(
    const float* __restrict__ qkv, const float* __restrict__ tWq,
    const float* __restrict__ tWv, const float* __restrict__ alpha,
    const int* __restrict__ pos, float* __restrict__ tauq,
    float* __restrict__ tauv)
{
  __shared__ float sfeat[3072];
  __shared__ float sred[256];
  const int s = blockIdx.x;
  const int tid = threadIdx.x;
#pragma unroll
  for (int i = 0; i < 12; ++i) {
    int idx = i * 256 + tid;
    float x = qkv[(size_t)s * 3072 + idx];
    sfeat[idx] = 0.5f * x * (1.0f + erff(x * 0.70710678118654752f)); // exact gelu
  }
  __syncthreads();
  const int o = tid >> 2;         // 0..63: 32 tok_q then 32 tok_v
  const int sub = tid & 3;
  const int hh = o & 31;
  const float* W = (o < 32) ? tWq : tWv;
  float a = 0.f;
  for (int i = sub; i < 3072; i += 4)
    a += sfeat[i] * W[i * 32 + hh];
  sred[tid] = a;
  __syncthreads();
  if (sub == 0) {
    float t = sred[tid] + sred[tid + 1] + sred[tid + 2] + sred[tid + 3];
    t = tanhf(t);
    float p = (float)pos[s] + 1.0f;
    float tp = 0.5f + 1.0f / (1.0f + __expf(-alpha[hh] * __logf(p)));
    float outv = t + tp;
    if (o < 32) tauq[s * 32 + hh] = outv;
    else        tauv[s * 32 + hh] = outv;
  }
}

// ---------------------------------------------------------------------------
// Apply tau scaling + partial RoPE; emit bf16 Q [H][S][64], K/V [KV][S][64].
// One block per sequence position (48 rows x 64 dims = qkv row).
// ---------------------------------------------------------------------------
__global__ __launch_bounds__(256) void qkv_post_kernel(
    const float* __restrict__ qkv, const float* __restrict__ tauq,
    const float* __restrict__ tauv, const float* __restrict__ cosp,
    const float* __restrict__ sinp,
    unsigned short* __restrict__ q_b, unsigned short* __restrict__ k_b,
    unsigned short* __restrict__ v_b)
{
  const int s = blockIdx.x;
  const int tid = threadIdx.x;
  const float* base = qkv + (size_t)s * 3072;
  const float* cs = cosp + s * 16;
  const float* sn = sinp + s * 16;
#pragma unroll
  for (int i = 0; i < 12; ++i) {
    int idx = i * 256 + tid;
    int row = idx >> 6;
    int d   = idx & 63;
    if (row < 32) {                       // Q heads: tau scale + rope
      const float* b = base + row * 64;
      float x;
      if (d < 32) {
        int ii = d >> 1;
        float xr = b[ii], xi = b[16 + ii];
        x = (d & 1) ? (xr * sn[ii] + xi * cs[ii]) : (xr * cs[ii] - xi * sn[ii]);
      } else x = b[d];
      x *= tauq[s * 32 + row];            // scalar tau commutes with rope
      q_b[((size_t)row * S_LEN + s) * 64 + d] = f32_to_bf16(x);
    } else if (row < 40) {                // K heads: rope only
      int kvh = row - 32;
      const float* b = base + 2048 + kvh * 64;
      float x;
      if (d < 32) {
        int ii = d >> 1;
        float xr = b[ii], xi = b[16 + ii];
        x = (d & 1) ? (xr * sn[ii] + xi * cs[ii]) : (xr * cs[ii] - xi * sn[ii]);
      } else x = b[d];
      k_b[((size_t)kvh * S_LEN + s) * 64 + d] = f32_to_bf16(x);
    } else {                              // V heads: tau_v scale
      int kvh = row - 40;
      float x = base[2560 + kvh * 64 + d] * tauv[s * 32 + kvh];
      v_b[((size_t)kvh * S_LEN + s) * 64 + d] = f32_to_bf16(x);
    }
  }
}

// ---------------------------------------------------------------------------
// Causal flash attention. One wave per (head, 32-query-row block): two 16-row
// M-tiles share every K/V fragment (2x arithmetic intensity on K/V traffic).
// Key chunks of 32: scores = 8 WMMAs, PV = 8 WMMAs per chunk. Online softmax
// with ds_swizzle xor-reductions within 16-lane halves (C-layout rows).
// ---------------------------------------------------------------------------
__global__ __launch_bounds__(256) void flash_attn_kernel(
    const unsigned short* __restrict__ qb,
    const unsigned short* __restrict__ kbuf,
    const unsigned short* __restrict__ vbuf,
    unsigned short* __restrict__ ob)
{
  __shared__ float pbuf[8][32 * 32];      // per-wave P scratch (C->A relayout)
  const int tid   = threadIdx.x;
  const int lane  = tid & 31;
  const int wslot = tid >> 5;
  const int gw    = blockIdx.x * 8 + wslot;
  const int h     = gw >> 6;              // 64 query blocks of 32 rows per head
  const int qp    = gw & 63;
  const int kv    = h >> 2;               // GQA: 4 q-heads per kv-head
  const int q0    = qp * 32;

  const int half = lane >> 4;
  const int ml   = lane & 15;
  const int kb   = half * 8;

  // Q fragments: two M-tiles x two 32-wide d-slices
  ABf aq[2][2];
#pragma unroll
  for (int mt = 0; mt < 2; ++mt) {
    const unsigned short* qrow =
        qb + ((size_t)h * S_LEN + q0 + mt * 16 + ml) * DHEAD;
#pragma unroll
    for (int r = 0; r < 8; ++r) {
      int k2 = pair_k(r, kb);
      aq[mt][0].u[r] = *(const unsigned int*)&qrow[k2];
      aq[mt][1].u[r] = *(const unsigned int*)&qrow[32 + k2];
    }
  }

  float m_i[2][8], l_i[2][8];
  v8f acc[2][4];
#pragma unroll
  for (int mt = 0; mt < 2; ++mt) {
#pragma unroll
    for (int r = 0; r < 8; ++r) { m_i[mt][r] = -__builtin_inff(); l_i[mt][r] = 0.f; }
#pragma unroll
    for (int t = 0; t < 4; ++t)
#pragma unroll
      for (int r = 0; r < 8; ++r) acc[mt][t][r] = 0.f;
  }

  const int kend = q0 + 32;               // causal: keys <= last query row
  for (int k0 = 0; k0 < kend; k0 += 32) {
    // K chunk B-fragments (shared by both M-tiles)
    ABf bk[2][2];                         // [key-tile][d-slice]
#pragma unroll
    for (int t = 0; t < 2; ++t) {
      const unsigned short* kr =
          kbuf + ((size_t)kv * S_LEN + k0 + t * 16 + ml) * DHEAD;
#pragma unroll
      for (int r = 0; r < 8; ++r) {
        int k2 = pair_k(r, kb);
        bk[t][0].u[r] = *(const unsigned int*)&kr[k2];
        bk[t][1].u[r] = *(const unsigned int*)&kr[32 + k2];
      }
    }
    // scores: 8 WMMAs
    v8f sc[2][2];
#pragma unroll
    for (int mt = 0; mt < 2; ++mt)
#pragma unroll
      for (int t = 0; t < 2; ++t) {
#pragma unroll
        for (int r = 0; r < 8; ++r) sc[mt][t][r] = 0.f;
        sc[mt][t] = wmma_bf16(aq[mt][0], bk[t][0], sc[mt][t]);
        sc[mt][t] = wmma_bf16(aq[mt][1], bk[t][1], sc[mt][t]);
      }

    // online softmax per M-tile
#pragma unroll
    for (int mt = 0; mt < 2; ++mt) {
#pragma unroll
      for (int r = 0; r < 8; ++r) {
        int qrow = q0 + mt * 16 + r + 8 * half;
        float s0 = ((k0 + ml)      <= qrow) ? sc[mt][0][r] * 0.125f
                                            : -__builtin_inff();
        float s1 = ((k0 + 16 + ml) <= qrow) ? sc[mt][1][r] * 0.125f
                                            : -__builtin_inff();
        float v  = half_max16(fmaxf(s0, s1));
        float nm   = fmaxf(m_i[mt][r], v);
        float corr = __expf(m_i[mt][r] - nm);
        float p0   = __expf(s0 - nm);
        float p1   = __expf(s1 - nm);
        float rs   = half_sum16(p0 + p1);
        l_i[mt][r] = l_i[mt][r] * corr + rs;
        m_i[mt][r] = nm;
        int prow = mt * 16 + r + 8 * half;
        pbuf[wslot][prow * 32 + ml]      = p0;
        pbuf[wslot][prow * 32 + 16 + ml] = p1;
#pragma unroll
        for (int t = 0; t < 4; ++t) acc[mt][t][r] *= corr;
      }
    }

    // P as 16x32 bf16 A-fragments (same-wave LDS bounce, DS ops in-order)
    ABf pa[2];
#pragma unroll
    for (int mt = 0; mt < 2; ++mt)
#pragma unroll
      for (int j = 0; j < 16; ++j)
        pa[mt].s[j] =
            f32_to_bf16(pbuf[wslot][(mt * 16 + ml) * 32 + elem_k(j, kb)]);

    // V chunk B-fragments (shared) + PV: 8 WMMAs
#pragma unroll
    for (int t = 0; t < 4; ++t) {
      ABf bv;
#pragma unroll
      for (int j = 0; j < 16; ++j) {
        int key = k0 + elem_k(j, kb);
        bv.s[j] = vbuf[((size_t)kv * S_LEN + key) * DHEAD + t * 16 + ml];
      }
      acc[0][t] = wmma_bf16(pa[0], bv, acc[0][t]);
      acc[1][t] = wmma_bf16(pa[1], bv, acc[1][t]);
    }
  }

  // epilogue
#pragma unroll
  for (int mt = 0; mt < 2; ++mt) {
    float inv[8];
#pragma unroll
    for (int r = 0; r < 8; ++r) inv[r] = 1.f / l_i[mt][r];
#pragma unroll
    for (int t = 0; t < 4; ++t)
#pragma unroll
      for (int r = 0; r < 8; ++r) {
        int row = q0 + mt * 16 + r + 8 * half;
        int d   = t * 16 + ml;
        ob[(size_t)row * (NH * DHEAD) + h * DHEAD + d] =
            f32_to_bf16(acc[mt][t][r] * inv[r]);
      }
  }
}

// ---------------------------------------------------------------------------
extern "C" void kernel_launch(void* const* d_in, const int* in_sizes, int n_in,
                              void* d_out, int out_size, void* d_ws,
                              size_t ws_size, hipStream_t stream) {
  (void)in_sizes; (void)n_in; (void)out_size; (void)ws_size;
  const float* hs   = (const float*)d_in[0];
  const float* cosp = (const float*)d_in[1];
  const float* sinp = (const float*)d_in[2];
  const int*   pos  = (const int*)d_in[3];
  const float* Wq   = (const float*)d_in[4];
  const float* Wk   = (const float*)d_in[5];
  const float* Wv   = (const float*)d_in[6];
  const float* Wo   = (const float*)d_in[7];
  const float* tWq  = (const float*)d_in[8];
  const float* tWv  = (const float*)d_in[9];
  const float* tAl  = (const float*)d_in[10];

  char* ws = (char*)d_ws;
  float* qkv          = (float*)(ws);                     // 2048*3072 f32 = 24 MiB
  float* tauq         = (float*)(ws + 25165824);          // 2048*32 f32
  float* tauv         = (float*)(ws + 25427968);          // 2048*32 f32
  unsigned short* q_b = (unsigned short*)(ws + 25690112); // 32*2048*64 bf16 = 8 MiB
  unsigned short* k_b = (unsigned short*)(ws + 34078720); // 8*2048*64 bf16 = 2 MiB
  unsigned short* v_b = (unsigned short*)(ws + 36175872); // 8*2048*64 bf16 = 2 MiB
  unsigned short* a_b = (unsigned short*)(ws + 38273024); // 2048*2048 bf16 = 8 MiB
  float* outp = (float*)d_out;

  // 1) QKV projection: hs(2048x2048) @ [Wq|Wk|Wv](2048x3072) -> qkv f32
  gemm_wmma<false><<<dim3(16, 24), 256, 0, stream>>>(
      hs, Wq, Wk, Wv, qkv, 2048, 3072, 2048, 2048, 2560, 2048, 512, 512);
  // 2) tau token/positional scalars
  tau_kernel<<<2048, 256, 0, stream>>>(qkv, tWq, tWv, tAl, pos, tauq, tauv);
  // 3) tau scaling + RoPE -> bf16 Q/K/V
  qkv_post_kernel<<<2048, 256, 0, stream>>>(qkv, tauq, tauv, cosp, sinp,
                                            q_b, k_b, v_b);
  // 4) causal flash attention -> bf16 attn output [S][H*DH]
  flash_attn_kernel<<<256, 256, 0, stream>>>(q_b, k_b, v_b, a_b);
  // 5) output projection: attn(2048x2048 bf16) @ Wo(2048x2048) -> d_out f32
  gemm_wmma<true><<<dim3(16, 16), 256, 0, stream>>>(
      a_b, Wo, Wo, Wo, outp, 2048, 2048, 2048, 2048, 2048, 2048, 2048, 2048);
}
